// CrossAttnFusion_71751723647635
// MI455X (gfx1250) — matrix operand order repaired and checked by reading, not compiled, not run on hardware
//
#include <hip/hip_runtime.h>
#include <math.h>

typedef __bf16 bf16_t;
typedef __attribute__((ext_vector_type(16))) __bf16 v16bf;
typedef __attribute__((ext_vector_type(8)))  __bf16 v8bf;
typedef __attribute__((ext_vector_type(8)))  float  v8f;

union AF { v16bf v; struct { v8bf lo, hi; } p; };

__device__ __forceinline__ v8f wmma_bf16(v16bf a, v16bf b, v8f c) {
  // D = A(16x32 bf16) * B(32x16 bf16) + C(16x16 f32)
  return __builtin_amdgcn_wmma_f32_16x16x32_bf16(
      /*neg_a=*/false, a, /*neg_b=*/false, b,
      /*c_mod=*/(short)0, c, /*reuse_a=*/false, /*reuse_b=*/false);
}

// ---------------------------------------------------------------------------
// fp32 -> bf16 conversion
// ---------------------------------------------------------------------------
__global__ void k_f32_to_bf16(const float* __restrict__ src,
                              bf16_t* __restrict__ dst, int n) {
  int i = blockIdx.x * blockDim.x + threadIdx.x;
  if (i < n) dst[i] = (bf16_t)src[i];
}

// ---------------------------------------------------------------------------
// C[M,N] = A[M,K](bf16) @ W[N,K]^T(bf16) + bias(f32)
// mode 0: out bf16 row-major (M,N)
// mode 1: out f32  row-major (M,N)
// mode 2: out bf16 V-transpose: m=b*512+t, n=h*64+d -> Vt[((b*16+h)*64+d)*512+t]
//
// Block: 256 threads = 8 waves. Block tile 64(M) x 128(N); each wave owns a
// 16-wide N strip and 4 M-tiles => every B-fragment is reused by 4 WMMAs with
// independent accumulators. All 4 A-fragments are loaded into distinct regs
// before the WMMA group so the 4 matrix ops issue back-to-back (single
// s_wait_loadcnt_dscnt per k-step instead of one wait per WMMA).
// ---------------------------------------------------------------------------
__global__ __launch_bounds__(256) void k_gemm_bf16(
    const bf16_t* __restrict__ A, const bf16_t* __restrict__ W,
    const float* __restrict__ bias, void* __restrict__ out,
    int M, int N, int K, int mode)
{
  __shared__ bf16_t sA[64][256];                  // 32 KB A chunk
  const int tid  = threadIdx.x;
  const int lane = tid & 31;
  const int wave = tid >> 5;
  const int m0   = blockIdx.y * 64;
  const int n0   = blockIdx.x * 128 + wave * 16;
  const int l15  = lane & 15;
  const int half = lane >> 4;
  const int kb   = half * 16;         // A-frag: lane<16 -> K{0..15}

  // B-frag: lane col = l15, K chunk = half*16 .. +15 (contiguous in W row)
  const bf16_t* wrow = W + (size_t)(n0 + l15) * K + half * 16;

  v8f acc[4] = {v8f{}, v8f{}, v8f{}, v8f{}};

  for (int kk = 0; kk < K; kk += 256) {
    __syncthreads();                  // protect sA against previous-chunk readers
    // cooperative load of 64x256 A chunk: 2048 x 16B, 8 chunks per thread
    #pragma unroll
    for (int i = 0; i < 8; ++i) {
      int c   = tid + 256 * i;
      int r   = c >> 5;               // 32 chunks of 8 halves per row
      int off = (c & 31) * 8;
      *(v8bf*)&sA[r][off] = *(const v8bf*)&A[(size_t)(m0 + r) * K + kk + off];
    }
    if (kk + 256 < K) __builtin_prefetch(wrow + kk + 256, 0, 1);
    __syncthreads();

    #pragma unroll
    for (int k0 = 0; k0 < 256; k0 += 32) {
      v16bf b = *(const v16bf*)(wrow + kk + k0);
      AF a0, a1, a2, a3;              // distinct regs: 8 ds_load_b128, one wait
      a0.p.lo = *(const v8bf*)&sA[ 0 + l15][k0 + kb];
      a0.p.hi = *(const v8bf*)&sA[ 0 + l15][k0 + kb + 8];
      a1.p.lo = *(const v8bf*)&sA[16 + l15][k0 + kb];
      a1.p.hi = *(const v8bf*)&sA[16 + l15][k0 + kb + 8];
      a2.p.lo = *(const v8bf*)&sA[32 + l15][k0 + kb];
      a2.p.hi = *(const v8bf*)&sA[32 + l15][k0 + kb + 8];
      a3.p.lo = *(const v8bf*)&sA[48 + l15][k0 + kb];
      a3.p.hi = *(const v8bf*)&sA[48 + l15][k0 + kb + 8];
      acc[0] = wmma_bf16(a0.v, b, acc[0]);
      acc[1] = wmma_bf16(a1.v, b, acc[1]);
      acc[2] = wmma_bf16(a2.v, b, acc[2]);
      acc[3] = wmma_bf16(a3.v, b, acc[3]);
    }
  }

  const int   col  = n0 + l15;
  const float bcol = bias[col];
  #pragma unroll
  for (int mt = 0; mt < 4; ++mt) {
    #pragma unroll
    for (int r = 0; r < 8; ++r) {
      int   m = m0 + mt * 16 + r + 8 * half;  // C layout: VGPR r -> M=r / 8+r
      float v = acc[mt][r] + bcol;
      if (mode == 0) {
        ((bf16_t*)out)[(size_t)m * N + col] = (bf16_t)v;
      } else if (mode == 1) {
        ((float*)out)[(size_t)m * N + col] = v;
      } else {
        int bb = m >> 9, t = m & 511, hh = col >> 6, dd = col & 63;
        ((bf16_t*)out)[(((size_t)bb * 16 + hh) * 64 + dd) * 512 + t] = (bf16_t)v;
      }
    }
  }
}

// ---------------------------------------------------------------------------
// Attention: one wave per (b, h, q-tile of 16). Scores 16x512 in LDS (fp32),
// softmax, then ctx(16x64) = P @ V via WMMA. Q/K: (rows,1024) bf16, V: (B,H,64,512).
// ---------------------------------------------------------------------------
__global__ __launch_bounds__(32) void k_attn(
    const bf16_t* __restrict__ Q,    // (B*256, 1024)
    const bf16_t* __restrict__ Kx,   // (B*512, 1024)
    const bf16_t* __restrict__ Vt,   // (B,H,64,512)
    bf16_t* __restrict__ ctx)        // (B*256, 1024)
{
  __shared__ float sc[16][512];      // 32 KB scores for this wave's q-tile
  const int id   = blockIdx.x;       // 4096 = 16 qt * 16 h * 16 b
  const int qt   = id & 15;
  const int h    = (id >> 4) & 15;
  const int b    = id >> 8;
  const int lane = threadIdx.x;
  const int l15  = lane & 15;
  const int half = lane >> 4;
  const int kb   = half * 16;

  // Q fragments: A 16x32 for d=0..31 and d=32..63
  const bf16_t* qrow = Q + (size_t)(b * 256 + qt * 16 + l15) * 1024 + h * 64;
  AF a0, a1;
  a0.p.lo = *(const v8bf*)(qrow + kb);
  a0.p.hi = *(const v8bf*)(qrow + kb + 8);
  a1.p.lo = *(const v8bf*)(qrow + 32 + kb);
  a1.p.hi = *(const v8bf*)(qrow + 32 + kb + 8);

  // scores = Q K^T / sqrt(64); B-frag lane: row k = l15, d chunk = half*16
  for (int kt = 0; kt < 32; ++kt) {
    const bf16_t* krow =
        Kx + (size_t)(b * 512 + kt * 16 + l15) * 1024 + h * 64 + half * 16;
    v16bf b0 = *(const v16bf*)krow;
    v16bf b1 = *(const v16bf*)(krow + 32);
    v8f s = {};
    s = wmma_bf16(a0.v, b0, s);
    s = wmma_bf16(a1.v, b1, s);
    #pragma unroll
    for (int r = 0; r < 8; ++r)
      sc[r + 8 * half][kt * 16 + l15] = s[r] * 0.125f;
  }
  __syncthreads();   // single-wave WG: NOP, kept for clarity

  // softmax over 512, two lanes per row
  {
    float* prow = &sc[l15][half * 256];
    float mx = -3.0e38f;
    for (int j = 0; j < 256; ++j) mx = fmaxf(mx, prow[j]);
    mx = fmaxf(mx, __shfl_xor(mx, 16, 32));
    float sum = 0.f;
    for (int j = 0; j < 256; ++j) {
      float e = __expf(prow[j] - mx);
      prow[j] = e;
      sum += e;
    }
    sum += __shfl_xor(sum, 16, 32);
    float inv = 1.f / sum;
    for (int j = 0; j < 256; ++j) prow[j] *= inv;
  }
  __syncthreads();

  // ctx(16x64) = P(16x512) @ V(512x64); V-frag contiguous thanks to Vt layout
  v8f acc[4] = {v8f{}, v8f{}, v8f{}, v8f{}};
  for (int kt = 0; kt < 16; ++kt) {
    // P fragment: two vectorized 32B LDS reads + packed f32->bf16 converts
    v8f plo = *(const v8f*)&sc[l15][kt * 32 + kb];
    v8f phi = *(const v8f*)&sc[l15][kt * 32 + kb + 8];
    AF p;
    #pragma unroll
    for (int j = 0; j < 8; ++j) {
      p.v[j]     = (bf16_t)plo[j];
      p.v[8 + j] = (bf16_t)phi[j];
    }
    const bf16_t* vbase =
        Vt + (((size_t)b * 16 + h) * 64) * 512 + kt * 32 + half * 16;
    #pragma unroll
    for (int nt = 0; nt < 4; ++nt) {
      v16bf bv = *(const v16bf*)(vbase + (size_t)(nt * 16 + l15) * 512);
      acc[nt] = wmma_bf16(p.v, bv, acc[nt]);
    }
  }

  #pragma unroll
  for (int nt = 0; nt < 4; ++nt) {
    #pragma unroll
    for (int r = 0; r < 8; ++r) {
      int m   = b * 256 + qt * 16 + r + 8 * half;
      int col = h * 64 + nt * 16 + l15;
      ctx[(size_t)m * 1024 + col] = (bf16_t)acc[nt][r];
    }
  }
}

// ---------------------------------------------------------------------------
// LayerNorm over D=1024 (population var), fp32 in -> bf16 out
// ---------------------------------------------------------------------------
__global__ __launch_bounds__(256) void k_layernorm(
    const float* __restrict__ x, const float* __restrict__ g,
    const float* __restrict__ bta, bf16_t* __restrict__ y, int D)
{
  __shared__ float r1[256], r2[256];
  const int row = blockIdx.x, t = threadIdx.x;
  const float* xr = x + (size_t)row * D;
  float s = 0.f, s2 = 0.f;
  for (int i = t; i < D; i += 256) { float v = xr[i]; s += v; s2 += v * v; }
  r1[t] = s; r2[t] = s2;
  __syncthreads();
  for (int o = 128; o > 0; o >>= 1) {
    if (t < o) { r1[t] += r1[t + o]; r2[t] += r2[t + o]; }
    __syncthreads();
  }
  float mu  = r1[0] / (float)D;
  float var = r2[0] / (float)D - mu * mu;
  float rs  = rsqrtf(var + 1e-5f);
  for (int i = t; i < D; i += 256)
    y[(size_t)row * D + i] = (bf16_t)((xr[i] - mu) * rs * g[i] + bta[i]);
}

// ---------------------------------------------------------------------------
// Segment mean: out[b,d] = mean over 256 rows of x[b*256+r, d]
// ---------------------------------------------------------------------------
__global__ void k_segmean(const float* __restrict__ x, float* __restrict__ out) {
  int idx = blockIdx.x * blockDim.x + threadIdx.x;   // 16384
  int b = idx >> 10, d = idx & 1023;
  const float* p = x + (size_t)b * 256 * 1024 + d;
  float s = 0.f;
  for (int r = 0; r < 256; ++r) s += p[(size_t)r * 1024];
  out[idx] = s * (1.0f / 256.0f);
}

// ---------------------------------------------------------------------------
extern "C" void kernel_launch(void* const* d_in, const int* in_sizes, int n_in,
                              void* d_out, int out_size, void* d_ws, size_t ws_size,
                              hipStream_t stream) {
  const float* struct_token = (const float*)d_in[0];   // (4096,1024)
  const float* text_token   = (const float*)d_in[1];   // (16,512,1024)
  const float* in_proj_w    = (const float*)d_in[2];   // (3072,1024)
  const float* in_proj_b    = (const float*)d_in[3];   // (3072,)
  const float* attn_out_w   = (const float*)d_in[4];   // (1024,1024)
  const float* attn_out_b   = (const float*)d_in[5];
  const float* ln_g         = (const float*)d_in[6];
  const float* ln_b         = (const float*)d_in[7];
  const float* proj_w       = (const float*)d_in[8];
  const float* proj_b       = (const float*)d_in[9];
  (void)in_sizes; (void)n_in; (void)out_size; (void)ws_size;

  char* ws = (char*)d_ws;
  const size_t MB = 1024u * 1024u;
  bf16_t* w_qkv_bf = (bf16_t*)(ws);             //  6 MB : wq|wk|wv bf16
  bf16_t* wo_bf    = (bf16_t*)(ws + 6 * MB);    //  2 MB
  bf16_t* wp_bf    = (bf16_t*)(ws + 8 * MB);    //  2 MB
  bf16_t* sbf      = (bf16_t*)(ws + 10 * MB);   //  8 MB struct bf16
  bf16_t* tbf      = (bf16_t*)(ws + 18 * MB);   // 16 MB text bf16
  bf16_t* Qbf      = (bf16_t*)(ws + 34 * MB);   //  8 MB
  bf16_t* Kbf      = (bf16_t*)(ws + 42 * MB);   // 16 MB
  bf16_t* Vtb      = (bf16_t*)(ws + 58 * MB);   // 16 MB (B,H,64,512)
  bf16_t* ctxbf    = (bf16_t*)(ws + 74 * MB);   //  8 MB
  float*  fusedf   = (float*) (ws + 82 * MB);   // 16 MB
  bf16_t* fusedbf  = (bf16_t*)(ws + 98 * MB);   //  8 MB
  float*  out2     = (float*) (ws + 106 * MB);  // 16 MB

  // conversions
  k_f32_to_bf16<<<(3 * 1024 * 1024 + 255) / 256, 256, 0, stream>>>(in_proj_w, w_qkv_bf, 3 * 1024 * 1024);
  k_f32_to_bf16<<<(1024 * 1024 + 255) / 256, 256, 0, stream>>>(attn_out_w, wo_bf, 1024 * 1024);
  k_f32_to_bf16<<<(1024 * 1024 + 255) / 256, 256, 0, stream>>>(proj_w, wp_bf, 1024 * 1024);
  k_f32_to_bf16<<<(4096 * 1024 + 255) / 256, 256, 0, stream>>>(struct_token, sbf, 4096 * 1024);
  k_f32_to_bf16<<<(8192 * 1024 + 255) / 256, 256, 0, stream>>>(text_token, tbf, 8192 * 1024);

  // Q = struct @ Wq^T   (4096 x 1024), bf16 out   — block tile 64x128
  k_gemm_bf16<<<dim3(8, 64), 256, 0, stream>>>(sbf, w_qkv_bf, in_proj_b, Qbf,
                                               4096, 1024, 1024, 0);
  // K = text @ Wk^T     (8192 x 1024), bf16 out
  k_gemm_bf16<<<dim3(8, 128), 256, 0, stream>>>(tbf, w_qkv_bf + (size_t)1024 * 1024,
                                                in_proj_b + 1024, Kbf,
                                                8192, 1024, 1024, 0);
  // V = text @ Wv^T, stored transposed (B,H,64,512)
  k_gemm_bf16<<<dim3(8, 128), 256, 0, stream>>>(tbf, w_qkv_bf + (size_t)2 * 1024 * 1024,
                                                in_proj_b + 2048, Vtb,
                                                8192, 1024, 1024, 2);
  // attention: 16 b * 16 h * 16 qtiles = 4096 waves
  k_attn<<<4096, 32, 0, stream>>>(Qbf, Kbf, Vtb, ctxbf);

  // fused = ctx @ Wo^T + b   (fp32 out)
  k_gemm_bf16<<<dim3(8, 64), 256, 0, stream>>>(ctxbf, wo_bf, attn_out_b, fusedf,
                                               4096, 1024, 1024, 1);
  // LayerNorm -> bf16
  k_layernorm<<<4096, 256, 0, stream>>>(fusedf, ln_g, ln_b, fusedbf, 1024);
  // out2 = fused @ Wp^T + b  (fp32 out)
  k_gemm_bf16<<<dim3(8, 64), 256, 0, stream>>>(fusedbf, wp_bf, proj_b, out2,
                                               4096, 1024, 1024, 1);
  // per-batch mean over 256 rows -> (16,1024)
  k_segmean<<<64, 256, 0, stream>>>(out2, (float*)d_out);
}
